// MoEFusionHead_21981642621064
// MI455X (gfx1250) — matrix operand order repaired and checked by reading, not compiled, not run on hardware
//
#include <hip/hip_runtime.h>
#include <hip/hip_bf16.h>
#include <math.h>

// Problem constants (match reference)
#define BB 4
#define SQL 1024
#define SKL 1024
#define DD  1024
#define HH  16
#define HDIM 64
#define EE  8
#define FFD 4096
#define TT  (BB*SQL)

typedef __bf16 bf16;
typedef __attribute__((ext_vector_type(16))) __bf16 v16bf;
typedef __attribute__((ext_vector_type(8)))  __bf16 v8bf;
typedef __attribute__((ext_vector_type(4)))  __bf16 v4bf;
typedef __attribute__((ext_vector_type(8)))  float  v8f;

// ---------------------------------------------------------------------------
// WMMA helpers (CDNA5 wave32, 16x16x32 bf16, fp32 accumulate)
// ---------------------------------------------------------------------------
static __device__ __forceinline__ v8f wmma_bf16(v16bf a, v16bf b, v8f c) {
  return __builtin_amdgcn_wmma_f32_16x16x32_bf16(
      /*neg_a=*/false, a, /*neg_b=*/false, b,
      /*c_mod=*/(short)0, c, /*reuse_a=*/false, /*reuse_b=*/false);
}

// A fragment (16x32 bf16) from row-major bf16 (LDS or global), row stride ld.
// ISA layout: lane m=lane&15; element i<8 -> k=hf*8+i, i>=8 -> k=16+hf*8+(i-8).
// Two contiguous 16B vector loads per lane.
static __device__ __forceinline__ v16bf load_A_bf(const bf16* __restrict__ src,
                                                  int ld, int lane) {
  const int m = lane & 15, hf = lane >> 4;
  const bf16* p = src + (size_t)m * ld + hf * 8;
  const v8bf lo = *(const v8bf*)(p);
  const v8bf hi = *(const v8bf*)(p + 16);
  v16bf a;
#pragma unroll
  for (int i = 0; i < 8; ++i) { a[i] = lo[i]; a[8 + i] = hi[i]; }
  return a;
}

// B fragment (32x16) from N-MAJOR bf16 source: src[n][k], row stride ld.
// ISA layout: lane n=lane&15 is the column; element i<8 -> k=hf*8+i, i>=8 -> k=16+hf*8+(i-8).
// Two contiguous 16B vector loads per lane.
static __device__ __forceinline__ v16bf load_B_bf(const bf16* __restrict__ src,
                                                  int ld, int lane) {
  const int n = lane & 15, hf = lane >> 4;
  const bf16* p = src + (size_t)n * ld + hf * 8;
  const v8bf lo = *(const v8bf*)(p);
  const v8bf hi = *(const v8bf*)(p + 16);
  v16bf b;
#pragma unroll
  for (int i = 0; i < 8; ++i) { b[i] = lo[i]; b[8 + i] = hi[i]; }
  return b;
}

static __device__ __forceinline__ float gelu_tanh(float x) {
  const float c = 0.79788456080286535588f;
  return 0.5f * x * (1.0f + tanhf(c * (x + 0.044715f * x * x * x)));
}

// ---------------------------------------------------------------------------
// Prep kernels: fp32 -> bf16 conversions / transposes (amortized per launch;
// bf16 expert weights total 128 MB -> fully L2-resident on MI455X's 192 MB L2)
// ---------------------------------------------------------------------------
__global__ __launch_bounds__(256) void k_transpose_bf16(
    const float* __restrict__ src, bf16* __restrict__ dst, int R, int C) {
  __shared__ float tile[32][33];
  const size_t mat = blockIdx.z;
  src += mat * (size_t)R * C;
  dst += mat * (size_t)R * C;
  const int c0 = blockIdx.x * 32, r0 = blockIdx.y * 32;
  const int tx = threadIdx.x & 31, ty = threadIdx.x >> 5;
#pragma unroll
  for (int i = 0; i < 32; i += 8)
    tile[ty + i][tx] = src[(size_t)(r0 + ty + i) * C + c0 + tx];
  __syncthreads();
#pragma unroll
  for (int i = 0; i < 32; i += 8)
    dst[(size_t)(c0 + ty + i) * R + r0 + tx] = (bf16)tile[tx][ty + i];
}

__global__ __launch_bounds__(256) void k_to_bf16(
    const float* __restrict__ s, bf16* __restrict__ d, int n4) {
  const int i = blockIdx.x * 256 + threadIdx.x;
  if (i < n4) {
    const float4 f = ((const float4*)s)[i];
    v4bf o;
    o[0] = (bf16)f.x; o[1] = (bf16)f.y; o[2] = (bf16)f.z; o[3] = (bf16)f.w;
    ((v4bf*)d)[i] = o;
  }
}

// v [B*SK, D] (head-interleaved) -> vt[b][h][hd][sk] bf16 (keys contiguous)
__global__ __launch_bounds__(256) void k_vtrans(
    const float* __restrict__ v, bf16* __restrict__ vt) {
  __shared__ float tile[32][33];
  const int bh = blockIdx.z;            // b*HH + h
  const int b = bh / HH, h = bh % HH;
  const int s0 = blockIdx.x * 32, d0 = blockIdx.y * 32;
  const int tx = threadIdx.x & 31, ty = threadIdx.x >> 5;
#pragma unroll
  for (int i = 0; i < 32; i += 8)
    tile[ty + i][tx] = v[(size_t)(b * SKL + s0 + ty + i) * DD + h * HDIM + d0 + tx];
  __syncthreads();
#pragma unroll
  for (int i = 0; i < 32; i += 8)
    vt[((size_t)bh * HDIM + d0 + ty + i) * SKL + s0 + tx] = (bf16)tile[tx][ty + i];
}

// ---------------------------------------------------------------------------
// Kernel 1: OUT[M,N] = A[M,K] @ W + bias, with W pre-transposed bf16 [N,K].
// Block = 8 waves, tile 16 x 256; A panel staged once in LDS as bf16.
// ---------------------------------------------------------------------------
__global__ __launch_bounds__(256) void k_gemm_bias(
    const float* __restrict__ A, const bf16* __restrict__ Wt,
    const float* __restrict__ bias, float* __restrict__ OUT,
    int M, int N, int K) {
  __shared__ bf16 alds[16 * DD];
  const int tid = threadIdx.x;
  const int m0 = blockIdx.y * 16;
  for (int idx = tid; idx < 16 * (K / 4); idx += 256) {
    const int r = idx / (K / 4), c4 = (idx % (K / 4)) * 4;
    const float4 f = *(const float4*)(A + (size_t)(m0 + r) * K + c4);
    alds[r * K + c4 + 0] = (bf16)f.x;
    alds[r * K + c4 + 1] = (bf16)f.y;
    alds[r * K + c4 + 2] = (bf16)f.z;
    alds[r * K + c4 + 3] = (bf16)f.w;
  }
  __syncthreads();
  const int w = tid >> 5, lane = tid & 31;
  const int hf = lane >> 4, nn = lane & 15;
  const int n0 = blockIdx.x * 256 + w * 32;
  const bf16* w0 = Wt + (size_t)n0 * K;        // fragment cols n0..n0+15
  const bf16* w1 = w0 + (size_t)16 * K;        // fragment cols n0+16..n0+31
  v8f acc0 = {}, acc1 = {};
  for (int kb = 0; kb < K; kb += 32) {
    const v16bf a  = load_A_bf(alds + kb, K, lane);
    const v16bf b0 = load_B_bf(w0 + kb, K, lane);
    const v16bf b1 = load_B_bf(w1 + kb, K, lane);
    acc0 = wmma_bf16(a, b0, acc0);
    acc1 = wmma_bf16(a, b1, acc1);
  }
#pragma unroll
  for (int r = 0; r < 8; ++r) {
    const size_t row = m0 + r + hf * 8;
    OUT[row * N + n0 + nn]      = acc0[r] + bias[n0 + nn];
    OUT[row * N + n0 + 16 + nn] = acc1[r] + bias[n0 + 16 + nn];
  }
}

// ---------------------------------------------------------------------------
// Kernel 2: flash attention, bf16 operands. One wave per (b, h, 16-row q tile).
// ---------------------------------------------------------------------------
__global__ __launch_bounds__(256) void k_attn(
    const bf16* __restrict__ qb, const bf16* __restrict__ kb,
    const bf16* __restrict__ vt, float* __restrict__ ctx) {
  __shared__ bf16 pbuf[8][16 * 32];
  const int tid = threadIdx.x, w = tid >> 5, lane = tid & 31;
  const int gt = blockIdx.x * 8 + w;
  const int qt = gt & 63;          // SQ/16 tiles
  const int h  = (gt >> 6) & 15;
  const int b  = gt >> 10;
  const int hf = lane >> 4, nn = lane & 15;
  const bf16* qbase = qb + ((size_t)(b * SQL + qt * 16)) * DD + h * HDIM;
  const bf16* kbase = kb + ((size_t)b * SKL) * DD + h * HDIM;
  const bf16* vbase = vt + ((size_t)(b * HH + h)) * HDIM * SKL;

  // Q fragments (A layout), two 16B loads per lane each
  const v16bf qa0 = load_A_bf(qbase, DD, lane);
  const v16bf qa1 = load_A_bf(qbase + 32, DD, lane);

  float rowmax[8], rowsum[8];
#pragma unroll
  for (int r = 0; r < 8; ++r) { rowmax[r] = -3.0e38f; rowsum[r] = 0.f; }
  v8f c0 = {}, c1 = {}, c2 = {}, c3 = {};
  bf16* pw = &pbuf[w][0];

  for (int kc = 0; kc < SKL; kc += 32) {
    // scores: B = K^T, K rows are n-major w.r.t. (k=hd, n=key) -> load_B_bf
    v8f s0 = {}, s1 = {};
    s0 = wmma_bf16(qa0, load_B_bf(kbase + (size_t)kc * DD, DD, lane), s0);
    s0 = wmma_bf16(qa1, load_B_bf(kbase + (size_t)kc * DD + 32, DD, lane), s0);
    s1 = wmma_bf16(qa0, load_B_bf(kbase + (size_t)(kc + 16) * DD, DD, lane), s1);
    s1 = wmma_bf16(qa1, load_B_bf(kbase + (size_t)(kc + 16) * DD + 32, DD, lane), s1);
#pragma unroll
    for (int r = 0; r < 8; ++r) {
      const float a0 = s0[r] * 0.125f, a1 = s1[r] * 0.125f;  // 1/sqrt(64)
      float mx = fmaxf(a0, a1);
      mx = fmaxf(mx, __shfl_xor(mx, 1, 32));
      mx = fmaxf(mx, __shfl_xor(mx, 2, 32));
      mx = fmaxf(mx, __shfl_xor(mx, 4, 32));
      mx = fmaxf(mx, __shfl_xor(mx, 8, 32));
      const float mnew = fmaxf(rowmax[r], mx);
      const float corr = __expf(rowmax[r] - mnew);
      rowmax[r] = mnew;
      const float p0 = __expf(a0 - mnew), p1 = __expf(a1 - mnew);
      float rs = p0 + p1;
      rs += __shfl_xor(rs, 1, 32); rs += __shfl_xor(rs, 2, 32);
      rs += __shfl_xor(rs, 4, 32); rs += __shfl_xor(rs, 8, 32);
      rowsum[r] = rowsum[r] * corr + rs;
      const int m = r + hf * 8;
      pw[m * 32 + nn]      = (bf16)p0;  // C layout -> A layout via LDS
      pw[m * 32 + 16 + nn] = (bf16)p1;
      c0[r] *= corr; c1[r] *= corr; c2[r] *= corr; c3[r] *= corr;
    }
    const v16bf pa = load_A_bf(pw, 32, lane);  // same-wave LDS RAW, hw-ordered
    // P@V: vt is n-major (n=hd rows, k=key contiguous)
    c0 = wmma_bf16(pa, load_B_bf(vbase + (size_t)0  * SKL + kc, SKL, lane), c0);
    c1 = wmma_bf16(pa, load_B_bf(vbase + (size_t)16 * SKL + kc, SKL, lane), c1);
    c2 = wmma_bf16(pa, load_B_bf(vbase + (size_t)32 * SKL + kc, SKL, lane), c2);
    c3 = wmma_bf16(pa, load_B_bf(vbase + (size_t)48 * SKL + kc, SKL, lane), c3);
  }
  float* obase = ctx + ((size_t)(b * SQL + qt * 16)) * DD + h * HDIM;
#pragma unroll
  for (int r = 0; r < 8; ++r) {
    const float inv = 1.0f / rowsum[r];
    const int m = r + hf * 8;
    obase[(size_t)m * DD + nn]      = c0[r] * inv;
    obase[(size_t)m * DD + 16 + nn] = c1[r] * inv;
    obase[(size_t)m * DD + 32 + nn] = c2[r] * inv;
    obase[(size_t)m * DD + 48 + nn] = c3[r] * inv;
  }
}

// ---------------------------------------------------------------------------
// Kernel 3: out[t] = LayerNorm(a[t] + b[t]) * g + beta.  One block per row.
// ---------------------------------------------------------------------------
__global__ __launch_bounds__(256) void k_add_ln(
    const float* __restrict__ a, const float* __restrict__ b,
    const float* __restrict__ g, const float* __restrict__ beta,
    float* __restrict__ out) {
  __shared__ float red[256];
  const int t = blockIdx.x, tid = threadIdx.x;
  const float* pa = a + (size_t)t * DD;
  const float* pb = b + (size_t)t * DD;
  float v[4], s = 0.f, s2 = 0.f;
#pragma unroll
  for (int i = 0; i < 4; ++i) {
    const int c = tid + i * 256;
    v[i] = pa[c] + pb[c];
    s += v[i]; s2 += v[i] * v[i];
  }
  red[tid] = s; __syncthreads();
  for (int off = 128; off > 0; off >>= 1) { if (tid < off) red[tid] += red[tid + off]; __syncthreads(); }
  const float mean = red[0] * (1.0f / DD); __syncthreads();
  red[tid] = s2; __syncthreads();
  for (int off = 128; off > 0; off >>= 1) { if (tid < off) red[tid] += red[tid + off]; __syncthreads(); }
  const float var = red[0] * (1.0f / DD) - mean * mean;
  const float inv = rsqrtf(var + 1e-5f);
#pragma unroll
  for (int i = 0; i < 4; ++i) {
    const int c = tid + i * 256;
    out[(size_t)t * DD + c] = (v[i] - mean) * inv * g[c] + beta[c];
  }
}

// ---------------------------------------------------------------------------
// Kernel 4: gate + top-2 routing, per-expert gather lists, aux-loss stats.
// ---------------------------------------------------------------------------
__global__ __launch_bounds__(256) void k_route(
    const float* __restrict__ x, const float* __restrict__ gw,
    int* __restrict__ counts, int* __restrict__ idxbuf,
    float* __restrict__ wbuf, float* __restrict__ imp,
    float* __restrict__ loadv) {
  __shared__ float lg[EE];
  const int t = blockIdx.x, tid = threadIdx.x;
  const int e = tid >> 5, lane = tid & 31;
  const float* xr = x + (size_t)t * DD;
  float s = 0.f;
  for (int c = lane; c < DD; c += 32) s += xr[c] * gw[(size_t)c * EE + e];
#pragma unroll
  for (int d = 16; d > 0; d >>= 1) s += __shfl_xor(s, d, 32);
  if (lane == 0) lg[e] = s;
  __syncthreads();
  if (tid == 0) {
    float mx = lg[0];
    for (int i = 1; i < EE; ++i) mx = fmaxf(mx, lg[i]);
    float p[EE], sum = 0.f;
    for (int i = 0; i < EE; ++i) { p[i] = __expf(lg[i] - mx); sum += p[i]; }
    const float inv = 1.f / sum;
    int i0 = 0;
    for (int i = 1; i < EE; ++i) if (p[i] > p[i0]) i0 = i;
    int i1 = (i0 == 0) ? 1 : 0;
    for (int i = 0; i < EE; ++i) if (i != i0 && p[i] > p[i1]) i1 = i;
    const float p0 = p[i0] * inv, p1 = p[i1] * inv;
    const float nw = 1.f / (p0 + p1);
    for (int i = 0; i < EE; ++i) atomicAdd(&imp[i], p[i] * inv * (1.0f / TT));
    atomicAdd(&loadv[i0], 1.0f / TT);
    int pos = atomicAdd(&counts[i0], 1);
    idxbuf[i0 * TT + pos] = t; wbuf[i0 * TT + pos] = p0 * nw;
    pos = atomicAdd(&counts[i1], 1);
    idxbuf[i1 * TT + pos] = t; wbuf[i1 * TT + pos] = p1 * nw;
  }
}

// ---------------------------------------------------------------------------
// Kernel 5: grouped MoE FFN, fully fused (x@W1 -> gelu -> @W2), per
// (expert, 16-token tile). Weights are pre-transposed bf16 (L2-resident).
// ---------------------------------------------------------------------------
__global__ __launch_bounds__(256) void k_moe(
    const float* __restrict__ x,
    const bf16* __restrict__ ew1t,   // [E][FF][D]  (n-major for phase 1)
    const float* __restrict__ eb1,
    const bf16* __restrict__ ew2t,   // [E][D][FF]  (n-major for phase 2)
    const float* __restrict__ eb2,
    const int* __restrict__ counts, const int* __restrict__ idxbuf,
    const float* __restrict__ wbuf, float* __restrict__ moe_out) {
  __shared__ bf16 xlds[16 * DD];     // 32 KB gathered activations
  __shared__ bf16 hlds[16 * 128];    // 4 KB hidden chunk
  __shared__ int   tok[16];
  __shared__ float wt[16];
  const int e = blockIdx.y;
  const int n = counts[e];
  const int t0 = blockIdx.x * 16;
  if (t0 >= n) return;
  const int tid = threadIdx.x;
  if (tid < 16) {
    const int row = t0 + tid;
    if (row < n) { tok[tid] = idxbuf[e * TT + row]; wt[tid] = wbuf[e * TT + row]; }
    else         { tok[tid] = idxbuf[e * TT];       wt[tid] = 0.f; }
  }
  __syncthreads();
  for (int idx = tid; idx < 16 * (DD / 4); idx += 256) {
    const int r = idx / (DD / 4), c4 = (idx % (DD / 4)) * 4;
    const float4 f = *(const float4*)(x + (size_t)tok[r] * DD + c4);
    xlds[r * DD + c4 + 0] = (bf16)f.x;
    xlds[r * DD + c4 + 1] = (bf16)f.y;
    xlds[r * DD + c4 + 2] = (bf16)f.z;
    xlds[r * DD + c4 + 3] = (bf16)f.w;
  }
  __syncthreads();
  const int w = tid >> 5, lane = tid & 31, hf = lane >> 4, nn = lane & 15;
  const bf16* W1 = ew1t + (size_t)e * DD * FFD;
  const bf16* W2 = ew2t + (size_t)e * FFD * DD;
  v8f oacc[8] = {};
  for (int fb = 0; fb < FFD; fb += 128) {
    // phase 1: h[16 x 16] for this wave's FF columns, K = 1024
    const int hn = fb + w * 16;
    const bf16* w1c = W1 + (size_t)hn * DD;
    v8f hacc = {};
    for (int kb = 0; kb < DD; kb += 32) {
      const v16bf a = load_A_bf(xlds + kb, DD, lane);
      const v16bf b = load_B_bf(w1c + kb, DD, lane);
      hacc = wmma_bf16(a, b, hacc);
    }
    if (fb + 128 < FFD) {  // hint next weight chunk toward L2 (global_prefetch_b8)
      __builtin_prefetch(W1 + (size_t)(fb + 128 + (tid >> 1)) * DD + (tid & 1) * 512, 0, 1);
      __builtin_prefetch(W2 + (size_t)(tid * 8) * FFD + fb + 128, 0, 1);
    }
#pragma unroll
    for (int r = 0; r < 8; ++r) {
      const int m = r + hf * 8;
      const float hv = gelu_tanh(hacc[r] + eb1[(size_t)e * FFD + hn + nn]);
      hlds[m * 128 + w * 16 + nn] = (bf16)hv;
    }
    __syncthreads();
    // phase 2: y[16 x 128 (this wave's D cols)] += h @ W2, K = 128
#pragma unroll
    for (int kb = 0; kb < 4; ++kb) {
      const v16bf a = load_A_bf(hlds + kb * 32, 128, lane);
#pragma unroll
      for (int nt = 0; nt < 8; ++nt) {
        const bf16* w2c = W2 + (size_t)(w * 128 + nt * 16) * FFD + fb + kb * 32;
        oacc[nt] = wmma_bf16(a, load_B_bf(w2c, FFD, lane), oacc[nt]);
      }
    }
    __syncthreads();
  }
#pragma unroll
  for (int nt = 0; nt < 8; ++nt) {
    const int col = w * 128 + nt * 16 + nn;
    const float b2 = eb2[(size_t)e * DD + col];
#pragma unroll
    for (int r = 0; r < 8; ++r) {
      const int m = r + hf * 8;
      atomicAdd(&moe_out[(size_t)tok[m] * DD + col], (oacc[nt][r] + b2) * wt[m]);
    }
  }
}

// ---------------------------------------------------------------------------
// Kernel 6: LN(x + moe) then mean-pool over sequence into pooled[B,D].
// ---------------------------------------------------------------------------
__global__ __launch_bounds__(256) void k_ln_pool(
    const float* __restrict__ x, const float* __restrict__ moe,
    const float* __restrict__ g, const float* __restrict__ beta,
    float* __restrict__ pooled) {
  __shared__ float red[256];
  const int t = blockIdx.x, tid = threadIdx.x;
  const int bidx = t / SQL;
  const float* pa = x + (size_t)t * DD;
  const float* pb = moe + (size_t)t * DD;
  float v[4], s = 0.f, s2 = 0.f;
#pragma unroll
  for (int i = 0; i < 4; ++i) {
    const int c = tid + i * 256;
    v[i] = pa[c] + pb[c];
    s += v[i]; s2 += v[i] * v[i];
  }
  red[tid] = s; __syncthreads();
  for (int off = 128; off > 0; off >>= 1) { if (tid < off) red[tid] += red[tid + off]; __syncthreads(); }
  const float mean = red[0] * (1.0f / DD); __syncthreads();
  red[tid] = s2; __syncthreads();
  for (int off = 128; off > 0; off >>= 1) { if (tid < off) red[tid] += red[tid + off]; __syncthreads(); }
  const float var = red[0] * (1.0f / DD) - mean * mean;
  const float inv = rsqrtf(var + 1e-5f);
#pragma unroll
  for (int i = 0; i < 4; ++i) {
    const int c = tid + i * 256;
    const float val = (v[i] - mean) * inv * g[c] + beta[c];
    atomicAdd(&pooled[(size_t)bidx * DD + c], val * (1.0f / SQL));
  }
}

__global__ void k_aux(const float* __restrict__ imp,
                      const float* __restrict__ loadv,
                      float* __restrict__ out) {
  if (threadIdx.x == 0) {
    float s = 0.f;
    for (int i = 0; i < EE; ++i) s += imp[i] * loadv[i];
    out[0] = (float)EE * s;
  }
}

// ---------------------------------------------------------------------------
extern "C" void kernel_launch(void* const* d_in, const int* in_sizes, int n_in,
                              void* d_out, int out_size, void* d_ws, size_t ws_size,
                              hipStream_t stream) {
  (void)in_sizes; (void)n_in; (void)out_size; (void)ws_size;
  const float* x1     = (const float*)d_in[0];
  const float* x2     = (const float*)d_in[1];
  const float* wq     = (const float*)d_in[2];
  const float* bq     = (const float*)d_in[3];
  const float* wk     = (const float*)d_in[4];
  const float* bk     = (const float*)d_in[5];
  const float* wv     = (const float*)d_in[6];
  const float* bv     = (const float*)d_in[7];
  const float* wo     = (const float*)d_in[8];
  const float* bo     = (const float*)d_in[9];
  const float* ln1_g  = (const float*)d_in[10];
  const float* ln1_b  = (const float*)d_in[11];
  const float* ln2_g  = (const float*)d_in[12];
  const float* ln2_b  = (const float*)d_in[13];
  const float* gate_w = (const float*)d_in[14];
  const float* ew1    = (const float*)d_in[15];
  const float* eb1    = (const float*)d_in[16];
  const float* ew2    = (const float*)d_in[17];
  const float* eb2    = (const float*)d_in[18];
  float* out = (float*)d_out;

  // workspace carving (activation buffers aliased across phases)
  char* ws = (char*)d_ws;
  size_t off = 0;
  auto carve = [&](size_t bytes) {
    char* p = ws + off;
    off += (bytes + 255) & ~(size_t)255;
    return (void*)p;
  };
  const size_t actB = sizeof(float) * (size_t)TT * DD;
  float* b0 = (float*)carve(actB);               // q -> attn_out
  float* b1 = (float*)carve(actB);               // k -> x (post-LN1)
  float* b2 = (float*)carve(actB);               // v -> moe_out
  float* b3 = (float*)carve(actB);               // ctx
  bf16* wqt  = (bf16*)carve(2ull * DD * DD);     // [N,K] bf16 weights
  bf16* wkt  = (bf16*)carve(2ull * DD * DD);
  bf16* wvt  = (bf16*)carve(2ull * DD * DD);
  bf16* wot  = (bf16*)carve(2ull * DD * DD);
  bf16* e1t  = (bf16*)carve(2ull * EE * DD * FFD);  // [E][FF][D]
  bf16* e2t  = (bf16*)carve(2ull * EE * FFD * DD);  // [E][D][FF]
  bf16* qbf  = (bf16*)carve(2ull * TT * DD);
  bf16* kbf  = (bf16*)carve(2ull * TT * DD);
  bf16* vtb  = (bf16*)carve(2ull * TT * DD);
  int*   counts = (int*)carve(sizeof(int) * EE * 3);
  float* imp    = (float*)(counts + EE);
  float* loadv  = imp + EE;
  int*   idxbuf = (int*)carve(sizeof(int) * EE * TT);
  float* wbuf   = (float*)carve(sizeof(float) * EE * TT);

  const dim3 blk(256);
  const dim3 gg(DD / 256, TT / 16);
  const dim3 gt(DD / 32, DD / 32, 1);

  // weight prep: fp32 -> bf16, N-major (one streaming pass, then L2-resident)
  k_transpose_bf16<<<gt, blk, 0, stream>>>(wq, wqt, DD, DD);
  k_transpose_bf16<<<gt, blk, 0, stream>>>(wk, wkt, DD, DD);
  k_transpose_bf16<<<gt, blk, 0, stream>>>(wv, wvt, DD, DD);
  k_transpose_bf16<<<gt, blk, 0, stream>>>(wo, wot, DD, DD);
  k_transpose_bf16<<<dim3(FFD / 32, DD / 32, EE), blk, 0, stream>>>(ew1, e1t, DD, FFD);
  k_transpose_bf16<<<dim3(DD / 32, FFD / 32, EE), blk, 0, stream>>>(ew2, e2t, FFD, DD);

  // QKV projections (+ bf16 operand prep for attention)
  k_gemm_bias<<<gg, blk, 0, stream>>>(x1, wqt, bq, b0, TT, DD, DD);
  k_gemm_bias<<<gg, blk, 0, stream>>>(x2, wkt, bk, b1, TT, DD, DD);
  k_gemm_bias<<<gg, blk, 0, stream>>>(x2, wvt, bv, b2, TT, DD, DD);
  k_to_bf16<<<dim3(TT * DD / 4 / 256), blk, 0, stream>>>(b0, qbf, TT * DD / 4);
  k_to_bf16<<<dim3(TT * DD / 4 / 256), blk, 0, stream>>>(b1, kbf, TT * DD / 4);
  k_vtrans<<<dim3(SKL / 32, HDIM / 32, BB * HH), blk, 0, stream>>>(b2, vtb);

  // attention + output projection + LN1
  k_attn<<<dim3(BB * HH * (SQL / 16) / 8), blk, 0, stream>>>(qbf, kbf, vtb, b3);
  k_gemm_bias<<<gg, blk, 0, stream>>>(b3, wot, bo, b0, TT, DD, DD);  // q dead
  k_add_ln<<<dim3(TT), blk, 0, stream>>>(x1, b0, ln1_g, ln1_b, b1);  // k dead

  // MoE: zero stats + accumulators (capture-safe memset nodes), route, FFN
  hipMemsetAsync(counts, 0, sizeof(int) * EE * 3, stream);
  hipMemsetAsync(b2, 0, actB, stream);                               // v dead
  hipMemsetAsync(out, 0, sizeof(float) * BB * DD, stream);
  k_route<<<dim3(TT), blk, 0, stream>>>(b1, gate_w, counts, idxbuf, wbuf, imp, loadv);
  k_moe<<<dim3(TT / 16, EE), blk, 0, stream>>>(b1, e1t, eb1, e2t, eb2,
                                               counts, idxbuf, wbuf, b2);

  // final LN + pool + aux loss
  k_ln_pool<<<dim3(TT), blk, 0, stream>>>(b1, b2, ln2_g, ln2_b, out);
  k_aux<<<dim3(1), dim3(32), 0, stream>>>(imp, loadv, out + BB * DD);
}
